// PhysicsGPT_69715909149470
// MI455X (gfx1250) — matrix operand order repaired and checked by reading, not compiled, not run on hardware
//
#include <hip/hip_runtime.h>
#include <hip/hip_bf16.h>
#include <math.h>

// ---------------- model dimensions ----------------
#define VCAB    32000
#define DMODEL  1024
#define NHEAD   16
#define NLAYER  6
#define TTEXT   768
#define KVIS    256
#define BATCH   2
#define SEQ     1024            // TTEXT + KVIS
#define DHEAD   64              // DMODEL / NHEAD
#define MROWS   2048            // BATCH * SEQ
#define PHYS_TOK_ID 5

typedef __attribute__((ext_vector_type(16))) _Float16 v16h;
typedef __attribute__((ext_vector_type(8)))  _Float16 v8h;
typedef __attribute__((ext_vector_type(4)))  _Float16 v4h;
typedef __attribute__((ext_vector_type(8)))  float    v8f;
typedef __attribute__((ext_vector_type(4)))  float    v4f;

// ---------------- WMMA helpers ----------------
__device__ __forceinline__ v8f wmma_f16(v16h a, v16h b, v8f c) {
  // v_wmma_f32_16x16x32_f16  (8-arg form, probe-confirmed)
  return __builtin_amdgcn_wmma_f32_16x16x32_f16(false, a, false, b, (short)0, c,
                                                false, false);
}

__device__ __forceinline__ v16h combine8(v8h lo, v8h hi) {
  v16h r;
#pragma unroll
  for (int i = 0; i < 8; ++i) { r[i] = lo[i]; r[i + 8] = hi[i]; }
  return r;
}

// A frag in this file's k order (B loads match): lane row = lane%16; 16 contiguous halves at
// K = 16*(lane/16) .. 16*(lane/16)+15.  rowk0 -> &A[m][k0] (row-major f16).
__device__ __forceinline__ v16h load_afrag(const _Float16* rowk0, int lane) {
  int base = (lane >> 4) << 4;
  v8h lo = *(const v8h*)(rowk0 + base);
  v8h hi = *(const v8h*)(rowk0 + base + 8);
  return combine8(lo, hi);
}

// ---------------- fp32 -> fp16 weight conversion (one pass, vectorized) ----------
__global__ void __launch_bounds__(256)
cvt_w16(const float* __restrict__ src, _Float16* __restrict__ dst, int n4) {
  int i = blockIdx.x * 256 + threadIdx.x;
  if (i >= n4) return;
  v4f v = ((const v4f*)src)[i];
  v4h h;
#pragma unroll
  for (int j = 0; j < 4; ++j) h[j] = (_Float16)v[j];
  ((v4h*)dst)[i] = h;
}

// ---------------- pe scan ----------------
__global__ void pe_kernel(const int* __restrict__ idx, int* __restrict__ pe) {
  int b = threadIdx.x;
  if (b >= BATCH) return;
  int p = 0;
  for (int t = 0; t < TTEXT; ++t) {
    if (idx[b * TTEXT + t] == PHYS_TOK_ID) { p = t + 1; break; }
  }
  pe[b] = p;
}

// ---------------- fused embedding ----------------
__global__ void __launch_bounds__(256)
embed_kernel(const int* __restrict__ idx, const float* __restrict__ vis,
             const float* __restrict__ wte, const float* __restrict__ wpe,
             const int* __restrict__ peArr, float* __restrict__ x) {
  int gid = blockIdx.x * blockDim.x + threadIdx.x;
  if (gid >= BATCH * SEQ * DMODEL) return;
  int d = gid % DMODEL;
  int p = (gid / DMODEL) % SEQ;
  int b = gid / (DMODEL * SEQ);
  int pe = peArr[b];
  float val;
  if (p >= pe && p < pe + KVIS) {
    int vi = p - pe;
    val = vis[((size_t)b * KVIS + vi) * DMODEL + d];
  } else {
    int ti = (p < pe) ? p : (p - KVIS);
    ti = ti < 0 ? 0 : (ti > TTEXT - 1 ? TTEXT - 1 : ti);
    int tok = idx[b * TTEXT + ti];
    val = wte[(size_t)tok * DMODEL + d];
  }
  x[gid] = val + wpe[(size_t)p * DMODEL + d];
}

// ---------------- layernorm (f32 in -> f16 out) ----------------
__global__ void __launch_bounds__(256)
ln_kernel(const float* __restrict__ x, const float* __restrict__ w,
          const float* __restrict__ bparm, _Float16* __restrict__ out) {
  __shared__ float rs[256], rs2[256];
  int row = blockIdx.x;
  int tid = threadIdx.x;
  const float* xr = x + (size_t)row * DMODEL;
  float s = 0.f, s2 = 0.f;
  for (int i = tid; i < DMODEL; i += 256) { float v = xr[i]; s += v; s2 += v * v; }
  rs[tid] = s; rs2[tid] = s2;
  __syncthreads();
  for (int off = 128; off > 0; off >>= 1) {
    if (tid < off) { rs[tid] += rs[tid + off]; rs2[tid] += rs2[tid + off]; }
    __syncthreads();
  }
  float mean = rs[0] * (1.0f / DMODEL);
  float var  = rs2[0] * (1.0f / DMODEL) - mean * mean;
  float rstd = rsqrtf(var + 1e-5f);
  for (int i = tid; i < DMODEL; i += 256)
    out[(size_t)row * DMODEL + i] = (_Float16)((xr[i] - mean) * rstd * w[i] + bparm[i]);
}

// ---------------- WMMA GEMM: C[M,N] = A[M,K](f16) * Wh[N,K](f16)^T ----------------
// Each wave owns a 32x64 tile: 2 A-frags x 4 B-frags -> 8 v_wmma per 32-K step.
// flags bit0 = exact GELU.
__global__ void __launch_bounds__(256)
gemm_wmma(const _Float16* __restrict__ A, const _Float16* __restrict__ Wh,
          const float* __restrict__ bias, const float* __restrict__ resid,
          float* __restrict__ outF, _Float16* __restrict__ outH,
          int Mm, int Nn, int Kk, int flags) {
  int wave = (blockIdx.x * blockDim.x + threadIdx.x) >> 5;
  int lane = threadIdx.x & 31;
  int tilesN = Nn >> 6;
  int total  = (Mm >> 5) * tilesN;
  if (wave >= total) return;                 // wave-uniform: EXEC stays all-ones
  int tm = wave / tilesN;
  int tn = wave - tm * tilesN;
  int m0 = tm << 5, n0 = tn << 6;
  int rc   = lane & 15;                      // A row / B col inside 16-tile
  int half = lane >> 4;
  const _Float16* Ap0 = A + (size_t)(m0 + rc) * Kk;
  const _Float16* Ap1 = A + (size_t)(m0 + 16 + rc) * Kk;
  const _Float16* Wp  = Wh + (size_t)(n0 + rc) * Kk + (half << 4);
  size_t ws16 = (size_t)16 * Kk;
  v8f z = {0,0,0,0,0,0,0,0};
  v8f c00 = z, c01 = z, c02 = z, c03 = z;    // rows m0..m0+15
  v8f c10 = z, c11 = z, c12 = z, c13 = z;    // rows m0+16..m0+31
  for (int k0 = 0; k0 < Kk; k0 += 32) {
    __builtin_prefetch((const void*)(Wp + k0 + 64), 0, 0);   // speculative: safe OOB
    v16h a0 = load_afrag(Ap0 + k0, lane);
    v16h a1 = load_afrag(Ap1 + k0, lane);
    v16h b0 = *(const v16h*)(Wp + k0);
    v16h b1 = *(const v16h*)(Wp + ws16 + k0);
    v16h b2 = *(const v16h*)(Wp + 2 * ws16 + k0);
    v16h b3 = *(const v16h*)(Wp + 3 * ws16 + k0);
    c00 = wmma_f16(a0, b0, c00);
    c10 = wmma_f16(a1, b0, c10);
    c01 = wmma_f16(a0, b1, c01);
    c11 = wmma_f16(a1, b1, c11);
    c02 = wmma_f16(a0, b2, c02);
    c12 = wmma_f16(a1, b2, c12);
    c03 = wmma_f16(a0, b3, c03);
    c13 = wmma_f16(a1, b3, c13);
  }
  v8f acc0[4] = {c00, c01, c02, c03};
  v8f acc1[4] = {c10, c11, c12, c13};
#pragma unroll
  for (int t = 0; t < 4; ++t) {
    int n = n0 + t * 16 + rc;
    float bv = bias ? bias[n] : 0.0f;
#pragma unroll
    for (int mi = 0; mi < 2; ++mi) {
      int mbase = m0 + mi * 16 + (half << 3);   // C/D: rows v + 8*(lane/16)
#pragma unroll
      for (int v = 0; v < 8; ++v) {
        float val = (mi == 0 ? acc0[t][v] : acc1[t][v]) + bv;
        if (flags & 1) val = 0.5f * val * (1.0f + erff(val * 0.70710678118f));
        size_t off = (size_t)(mbase + v) * Nn + n;
        if (resid) val += resid[off];
        if (outF) outF[off] = val;
        if (outH) outH[off] = (_Float16)val;
      }
    }
  }
}

// ---------------- split heads: qkv f32 -> q16/k16 (b,h,s,dh), v16t (b,h,dh,s) ----------------
__global__ void __launch_bounds__(256)
split_heads(const float* __restrict__ qkv, _Float16* __restrict__ q16,
            _Float16* __restrict__ k16, _Float16* __restrict__ v16t) {
  int gid = blockIdx.x * blockDim.x + threadIdx.x;
  if (gid >= BATCH * SEQ * DMODEL) return;
  int d = gid % DMODEL;
  int s = (gid / DMODEL) % SEQ;
  int b = gid / (DMODEL * SEQ);
  int h = d / DHEAD, dh = d % DHEAD;
  size_t rbase = (size_t)(b * SEQ + s) * (3 * DMODEL);
  float qv = qkv[rbase + d] * 0.125f;                 // fold 1/sqrt(DH)=0.125 into q
  float kv = qkv[rbase + DMODEL + d];
  float vv = qkv[rbase + 2 * DMODEL + d];
  size_t bh = (size_t)(b * NHEAD + h);
  q16[(bh * SEQ + s) * DHEAD + dh]  = (_Float16)qv;
  k16[(bh * SEQ + s) * DHEAD + dh]  = (_Float16)kv;
  v16t[(bh * DHEAD + dh) * SEQ + s] = (_Float16)vv;   // transposed for B fragments
}

// ---------------- flash-attention with WMMA (4 waves/block, 1 qtile/wave) ----------------
__global__ void __launch_bounds__(128)
attn_wmma(const _Float16* __restrict__ q16, const _Float16* __restrict__ k16,
          const _Float16* __restrict__ v16t, const int* __restrict__ peArr,
          _Float16* __restrict__ attnOut) {
  __shared__ __align__(32) _Float16 pbuf[4][16 * 32];
  int wInB = threadIdx.x >> 5;
  int lane = threadIdx.x & 31;
  int tile = blockIdx.x * 4 + wInB;          // (b*NHEAD+h)*(SEQ/16) + qt
  int qt = tile & (SEQ / 16 - 1);
  int bh = tile >> 6;                        // SEQ/16 == 64
  int b  = bh >> 4;                          // NHEAD == 16
  int h  = bh & 15;
  int pe = peArr[b];
  int ve = pe + KVIS;
  int rc   = lane & 15;
  int half = lane >> 4;

  const _Float16* Qp = q16 + ((size_t)bh * SEQ + qt * 16 + rc) * DHEAD;
  v16h aq0 = load_afrag(Qp, lane);           // dh 0..31
  v16h aq1 = load_afrag(Qp + 32, lane);      // dh 32..63
  const _Float16* Kc = k16 + (size_t)bh * SEQ * DHEAD;
  const _Float16* Vt = v16t + (size_t)bh * DHEAD * SEQ;

  v8f zero = {0,0,0,0,0,0,0,0};
  v8f o0 = zero, o1 = zero, o2 = zero, o3 = zero;
  float mst[8], lst[8];
#pragma unroll
  for (int v = 0; v < 8; ++v) { mst[v] = -3.0e30f; lst[v] = 0.0f; }
  int qbase = qt * 16 + half * 8;
  _Float16* pb = pbuf[wInB];

  for (int kb = 0; kb < SEQ; kb += 32) {
    v8f s0 = zero, s1 = zero;
    {
      const _Float16* Kp = Kc + (size_t)(kb + rc) * DHEAD + (half << 4);
      v16h bk0 = *(const v16h*)(Kp);
      v16h bk1 = *(const v16h*)(Kp + 32);
      s0 = wmma_f16(aq0, bk0, s0);
      s0 = wmma_f16(aq1, bk1, s0);
    }
    {
      const _Float16* Kp = Kc + (size_t)(kb + 16 + rc) * DHEAD + (half << 4);
      v16h bk0 = *(const v16h*)(Kp);
      v16h bk1 = *(const v16h*)(Kp + 32);
      s1 = wmma_f16(aq0, bk0, s1);
      s1 = wmma_f16(aq1, bk1, s1);
    }
    int c0i = kb + rc;
    int c1i = c0i + 16;
#pragma unroll
    for (int v = 0; v < 8; ++v) {
      int q = qbase + v;
      bool a0 = (q < pe && c0i < pe) || (q >= pe && q < ve && c0i < ve) ||
                (q >= ve && c0i <= q);
      bool a1 = (q < pe && c1i < pe) || (q >= pe && q < ve && c1i < ve) ||
                (q >= ve && c1i <= q);
      float sv0 = s0[v] + (a0 ? 0.0f : -1.0e9f);
      float sv1 = s1[v] + (a1 ? 0.0f : -1.0e9f);
      float mv = fmaxf(sv0, sv1);
#pragma unroll
      for (int off = 1; off < 16; off <<= 1) mv = fmaxf(mv, __shfl_xor(mv, off, 16));
      float mnew  = fmaxf(mst[v], mv);
      float scale = __expf(mst[v] - mnew);
      float p0 = __expf(sv0 - mnew);
      float p1 = __expf(sv1 - mnew);
      float ls = p0 + p1;
#pragma unroll
      for (int off = 1; off < 16; off <<= 1) ls += __shfl_xor(ls, off, 16);
      mst[v] = mnew;
      lst[v] = lst[v] * scale + ls;
      o0[v] *= scale; o1[v] *= scale; o2[v] *= scale; o3[v] *= scale;
      int r = half * 8 + v;
      pb[r * 32 + rc]      = (_Float16)p0;
      pb[r * 32 + rc + 16] = (_Float16)p1;
    }
    __syncthreads();
    v16h pa = load_afrag(pb + rc * 32, lane);   // P tile as A fragment via LDS
    __syncthreads();
#pragma unroll
    for (int t = 0; t < 4; ++t) {
      const _Float16* Vp = Vt + (size_t)(t * 16 + rc) * SEQ + kb + (half << 4);
      v16h bv = *(const v16h*)Vp;
      v8f& o = (t == 0 ? o0 : (t == 1 ? o1 : (t == 2 ? o2 : o3)));
      o = wmma_f16(pa, bv, o);
    }
  }

#pragma unroll
  for (int v = 0; v < 8; ++v) {
    float inv = 1.0f / lst[v];
    int m = qt * 16 + half * 8 + v;
    size_t rowoff = ((size_t)(b * SEQ + m)) * DMODEL + h * DHEAD;
#pragma unroll
    for (int t = 0; t < 4; ++t) {
      v8f& o = (t == 0 ? o0 : (t == 1 ? o1 : (t == 2 ? o2 : o3)));
      attnOut[rowoff + t * 16 + rc] = (_Float16)(o[v] * inv);
    }
  }
}

// ---------------- host orchestration ----------------
static inline void launch_gemm(const _Float16* A, const _Float16* Wh, const float* bias,
                               const float* resid, float* outF, _Float16* outH,
                               int Mm, int Nn, int Kk, int flags, hipStream_t s) {
  int tiles  = (Mm / 32) * (Nn / 64);
  int blocks = (tiles + 7) / 8;              // 8 waves per 256-thread block
  gemm_wmma<<<blocks, 256, 0, s>>>(A, Wh, bias, resid, outF, outH, Mm, Nn, Kk, flags);
}

static inline void launch_cvt(const float* src, _Float16* dst, size_t n, hipStream_t s) {
  int n4 = (int)(n / 4);
  cvt_w16<<<(n4 + 255) / 256, 256, 0, s>>>(src, dst, n4);
}

extern "C" void kernel_launch(void* const* d_in, const int* in_sizes, int n_in,
                              void* d_out, int out_size, void* d_ws, size_t ws_size,
                              hipStream_t stream) {
  (void)in_sizes; (void)n_in; (void)out_size; (void)ws_size;
  const int*   idx   = (const int*)  d_in[0];
  const float* vis   = (const float*)d_in[1];
  const float* wte   = (const float*)d_in[2];
  const float* wpe   = (const float*)d_in[3];
  const float* qkv_w = (const float*)d_in[4];
  const float* qkv_b = (const float*)d_in[5];
  const float* out_w = (const float*)d_in[6];
  const float* out_b = (const float*)d_in[7];
  const float* ln1_w = (const float*)d_in[8];
  const float* ln1_b = (const float*)d_in[9];
  const float* ln2_w = (const float*)d_in[10];
  const float* ln2_b = (const float*)d_in[11];
  const float* ff1_w = (const float*)d_in[12];
  const float* ff1_b = (const float*)d_in[13];
  const float* ff2_w = (const float*)d_in[14];
  const float* ff2_b = (const float*)d_in[15];
  const float* lnf_w = (const float*)d_in[16];
  const float* lnf_b = (const float*)d_in[17];

  char* wsp = (char*)d_ws;
  auto alloc = [&](size_t bytes) -> void* {
    void* p = (void*)wsp;
    wsp += (bytes + 255) & ~(size_t)255;
    return p;
  };
  int*       pe     = (int*)      alloc(BATCH * sizeof(int));
  float*     x      = (float*)    alloc((size_t)MROWS * DMODEL * 4);       // residual stream
  _Float16*  h16    = (_Float16*) alloc((size_t)MROWS * DMODEL * 2);       // LN output (A matrix)
  float*     qkv    = (float*)    alloc((size_t)MROWS * 3 * DMODEL * 4);
  _Float16*  q16    = (_Float16*) alloc((size_t)MROWS * DMODEL * 2);
  _Float16*  k16    = (_Float16*) alloc((size_t)MROWS * DMODEL * 2);
  _Float16*  v16t   = (_Float16*) alloc((size_t)MROWS * DMODEL * 2);
  _Float16*  attn16 = (_Float16*) alloc((size_t)MROWS * DMODEL * 2);
  _Float16*  ff16   = (_Float16*) alloc((size_t)MROWS * 4 * DMODEL * 2);
  // per-layer f16 weight slab (reused across layers)
  _Float16*  qkvw16 = (_Float16*) alloc((size_t)3 * DMODEL * DMODEL * 2);
  _Float16*  outw16 = (_Float16*) alloc((size_t)DMODEL * DMODEL * 2);
  _Float16*  ff1w16 = (_Float16*) alloc((size_t)4 * DMODEL * DMODEL * 2);
  _Float16*  ff2w16 = (_Float16*) alloc((size_t)4 * DMODEL * DMODEL * 2);
  _Float16*  wte16  = (_Float16*) alloc((size_t)VCAB * DMODEL * 2);

  const int elemBlocks = (BATCH * SEQ * DMODEL) / 256;   // 8192

  pe_kernel<<<1, 32, 0, stream>>>(idx, pe);
  embed_kernel<<<elemBlocks, 256, 0, stream>>>(idx, vis, wte, wpe, pe, x);
  launch_cvt(wte, wte16, (size_t)VCAB * DMODEL, stream);   // once per launch

  for (int l = 0; l < NLAYER; ++l) {
    // stage this layer's weights as f16
    launch_cvt(qkv_w + (size_t)l * 3 * DMODEL * DMODEL, qkvw16,
               (size_t)3 * DMODEL * DMODEL, stream);
    launch_cvt(out_w + (size_t)l * DMODEL * DMODEL, outw16,
               (size_t)DMODEL * DMODEL, stream);
    launch_cvt(ff1_w + (size_t)l * 4 * DMODEL * DMODEL, ff1w16,
               (size_t)4 * DMODEL * DMODEL, stream);
    launch_cvt(ff2_w + (size_t)l * 4 * DMODEL * DMODEL, ff2w16,
               (size_t)4 * DMODEL * DMODEL, stream);

    ln_kernel<<<MROWS, 256, 0, stream>>>(x, ln1_w + (size_t)l * DMODEL,
                                         ln1_b + (size_t)l * DMODEL, h16);
    launch_gemm(h16, qkvw16, qkv_b + (size_t)l * 3 * DMODEL, nullptr, qkv, nullptr,
                MROWS, 3 * DMODEL, DMODEL, 0, stream);
    split_heads<<<elemBlocks, 256, 0, stream>>>(qkv, q16, k16, v16t);
    attn_wmma<<<(BATCH * NHEAD * (SEQ / 16)) / 4, 128, 0, stream>>>(q16, k16, v16t,
                                                                    pe, attn16);
    launch_gemm(attn16, outw16, out_b + (size_t)l * DMODEL, x, x, nullptr,
                MROWS, DMODEL, DMODEL, 0, stream);
    ln_kernel<<<MROWS, 256, 0, stream>>>(x, ln2_w + (size_t)l * DMODEL,
                                         ln2_b + (size_t)l * DMODEL, h16);
    launch_gemm(h16, ff1w16, ff1_b + (size_t)l * 4 * DMODEL, nullptr, nullptr, ff16,
                MROWS, 4 * DMODEL, DMODEL, 1 /*gelu*/, stream);
    launch_gemm(ff16, ff2w16, ff2_b + (size_t)l * DMODEL, x, x, nullptr,
                MROWS, DMODEL, 4 * DMODEL, 0, stream);
  }

  ln_kernel<<<MROWS, 256, 0, stream>>>(x, lnf_w, lnf_b, h16);
  launch_gemm(h16, wte16, nullptr, nullptr, (float*)d_out, nullptr,
              MROWS, VCAB, DMODEL, 0, stream);
}
